// AMPBlock1_65128884076683
// MI455X (gfx1250) — matrix operand order repaired and checked by reading, not compiled, not run on hardware
//
#include <hip/hip_runtime.h>
#include <hip/hip_bf16.h>

typedef __bf16 v16bf __attribute__((ext_vector_type(16)));
typedef __bf16 v8bf  __attribute__((ext_vector_type(8)));
typedef float  v8f   __attribute__((ext_vector_type(8)));
typedef int    v4i   __attribute__((ext_vector_type(4)));

typedef __attribute__((address_space(1))) v4i* as1_v4i_ptr;
typedef __attribute__((address_space(3))) v4i* as3_v4i_ptr;

// Fixed problem shape (reference setup_inputs): B=4, T=8192, C=512, K=3.
#define BATCH 4
#define TLEN  8192
#define TSHIFT 13
#define TMASK 8191
#define CH    512
#define KTAP  3

// 12-tap Kaiser-sinc lowpass (cutoff=0.25, half_width=0.3, beta=4.6638), normalized.
__device__ __constant__ float FILT[12] = {
     0.0020289f,  0.0093898f, -0.0255440f, -0.0576500f,  0.1285733f,  0.4432020f,
     0.4432020f,  0.1285733f, -0.0576500f, -0.0255440f,  0.0093898f,  0.0020289f
};

// ---------------------------------------------------------------------------
// CDNA5 async global->LDS staging (ASYNCcnt), with safe fallback.
// ---------------------------------------------------------------------------
__device__ __forceinline__ void async_b128(const __bf16* gsrc, __bf16* ldst) {
#if __has_builtin(__builtin_amdgcn_global_load_async_to_lds_b128)
    __builtin_amdgcn_global_load_async_to_lds_b128(
        (as1_v4i_ptr)(uintptr_t)gsrc,
        (as3_v4i_ptr)(unsigned)(uintptr_t)ldst,
        0, 0);
#else
    *reinterpret_cast<uint4*>(ldst) = *reinterpret_cast<const uint4*>(gsrc);
#endif
}

// Wait until at most N async ops outstanding (ASYNCcnt credit semantics).
__device__ __forceinline__ void wait_async_le4() {
#if __has_builtin(__builtin_amdgcn_s_wait_asynccnt)
    __builtin_amdgcn_s_wait_asynccnt(4);
#else
    asm volatile("s_wait_asynccnt 0x4" ::: "memory");
#endif
}
__device__ __forceinline__ void wait_async_le0() {
#if __has_builtin(__builtin_amdgcn_s_wait_asynccnt)
    __builtin_amdgcn_s_wait_asynccnt(0);
#else
    asm volatile("s_wait_asynccnt 0x0" ::: "memory");
#endif
}

// ---------------------------------------------------------------------------
// Weight-norm fold, TRANSPOSED output layout for WMMA B-fragments:
//   wout[tap][co][ci] = bf16( v[tap][ci][co] / ||v[:,:,co]|| * g[co] )
// One block per output channel co.
// ---------------------------------------------------------------------------
__global__ __launch_bounds__(256)
void wnorm_kernel(const float* __restrict__ v, const float* __restrict__ g,
                  __bf16* __restrict__ wout) {
    int co  = blockIdx.x;
    int tid = threadIdx.x;
    __shared__ float red[256];
    const int n = KTAP * CH;
    float s = 0.f;
    for (int i = tid; i < n; i += 256) {
        float val = v[(long)i * CH + co];
        s += val * val;
    }
    red[tid] = s;
    __syncthreads();
    for (int st = 128; st > 0; st >>= 1) {
        if (tid < st) red[tid] += red[tid + st];
        __syncthreads();
    }
    float scale = g[co] * rsqrtf(red[0]);
    for (int i = tid; i < n; i += 256) {
        int k  = i >> 9;          // i / CH
        int ci = i & (CH - 1);    // i % CH
        wout[((long)(k * CH + co)) * CH + ci] = (__bf16)(v[(long)i * CH + co] * scale);
    }
}

// ---------------------------------------------------------------------------
// Fused anti-aliased snake: y = downsample2(snake(upsample2(x))), bf16 output.
//   u[2t+q] = 2 * sum_{k : (q+4+k) even} FILT[k] * x[clamp(t + (q+4+k)/2 - 5)]
//   y[t]    = sum_{q=-5..6} FILT[q+5] * snake(u[2t+q])
// ---------------------------------------------------------------------------
__global__ __launch_bounds__(256)
void act1d_kernel(const float* __restrict__ x, const float* __restrict__ log_alpha,
                  __bf16* __restrict__ out) {
    long idx = (long)blockIdx.x * blockDim.x + threadIdx.x;
    int c = (int)(idx & (CH - 1));
    int t = (int)((idx >> 9) & TMASK);
    int b = (int)(idx >> (9 + TSHIFT));

    float alpha = __expf(log_alpha[c]);
    float inv_a = 1.f / (alpha + 1e-9f);

    const float* xb = x + ((long)b << (9 + TSHIFT)) + c;
    float xv[11];
#pragma unroll
    for (int i = 0; i < 11; ++i) {
        int tt = t - 5 + i;
        tt = tt < 0 ? 0 : (tt > TLEN - 1 ? TLEN - 1 : tt);
        xv[i] = xb[(long)tt * CH];
    }

    float acc = 0.f;
#pragma unroll
    for (int q = -5; q <= 6; ++q) {
        float u = 0.f;
#pragma unroll
        for (int k = 0; k < 12; ++k) {
            int s = q + 4 + k;
            if ((s & 1) == 0) u += FILT[k] * xv[s >> 1];
        }
        u *= 2.f;
        float sn = __sinf(alpha * u);
        float sv = u + inv_a * sn * sn;
        acc += FILT[q + 5] * sv;
    }
    out[idx] = (__bf16)acc;
}

// ---------------------------------------------------------------------------
// Implicit-GEMM dilated conv (SAME, K=3) via v_wmma_f32_16x16x32_bf16.
// Block tile 128(M) x 128(N); 8 waves as 4(M) x 2(N); wave = 32x64 = 2x4 WMMAs.
// 48-stage K pipeline (3 taps x 16 kc-steps), double-buffered LDS tiles staged
// with GLOBAL_LOAD_ASYNC_TO_LDS_B128; s_wait_asynccnt<=4 overlaps stage s+1's
// HBM->LDS traffic with stage s's WMMA burst. Rows padded to 40 bf16 (80B) so
// fragment b128 loads are bank-conflict-free.
// ---------------------------------------------------------------------------
#define TM 128
#define TN 128
#define LDP 40
#define NSTAGE 48

__global__ __launch_bounds__(256)
void conv_wmma_kernel(const __bf16* __restrict__ Ag, const __bf16* __restrict__ Wt,
                      const float* __restrict__ bias, const float* __restrict__ resid,
                      float* __restrict__ out, int dil) {
    __shared__ alignas(16) __bf16 Asm[2][TM][LDP];
    __shared__ alignas(16) __bf16 Bsm[2][TN][LDP];

    int tid   = threadIdx.x;
    int lane  = tid & 31;
    int wave  = tid >> 5;
    int waveM = wave & 3;   // 0..3 -> 32-row slabs
    int waveN = wave >> 2;  // 0..1 -> 64-col slabs
    int m0 = blockIdx.x * TM;
    int n0 = blockIdx.y * TN;

    v8f acc[2][4] = {};

    // Per-thread staging coordinates (A): row r, 16-element half h.
    const int r  = tid >> 1;
    const int h  = tid & 1;
    const int m  = m0 + r;
    const int tt = m & TMASK;
    const long brow = (long)(m >> TSHIFT) << TSHIFT;   // batch base row

    // Stage s (tap = s>>4, kc = (s&15)*32) into LDS buffer `buf`.
    // Every wave issues exactly 4 async instructions per stage (2 A + 2 W):
    // at batch-boundary tiles the A-asyncs run with partial EXEC (at most 5
    // of a wave's 16 rows can be out of range, never all), and invalid rows
    // are zero-filled via DS stores (covered by the barrier).
    auto stage = [&](int s, int buf) {
        int tap = s >> 4;
        int kc  = (s & 15) << 5;
        int ti  = tt + (tap - 1) * dil;
        bool valid = (ti >= 0) && (ti < TLEN);
        __bf16* ldst = &Asm[buf][r][h * 16];
        if (valid) {
            const __bf16* gsrc = Ag + (brow + ti) * CH + kc + h * 16;
            async_b128(gsrc, ldst);
            async_b128(gsrc + 8, ldst + 8);
        } else {
            uint4 z = {0u, 0u, 0u, 0u};
            reinterpret_cast<uint4*>(ldst)[0] = z;
            *reinterpret_cast<uint4*>(ldst + 8) = z;
        }
#pragma unroll
        for (int j = 0; j < 2; ++j) {
            int chunk = tid * 2 + j;        // 0..511
            int row   = chunk >> 2;         // co local 0..127
            int seg   = chunk & 3;          // 8-element ci chunk
            const __bf16* gsrc =
                Wt + ((long)(tap * CH + n0 + row)) * CH + kc + seg * 8;
            async_b128(gsrc, &Bsm[buf][row][seg * 8]);
        }
    };

    const int lrow  = lane & 15;
    const int khalf = (lane >> 4) * 8;   // A: K octet select
    const int kb    = (lane >> 4) * 8;   // B: K octet select

    stage(0, 0);
    for (int s = 0; s < NSTAGE; ++s) {
        int cur = s & 1;
        if (s + 1 < NSTAGE) {
            stage(s + 1, cur ^ 1);
            wait_async_le4();   // stage s resident; stage s+1 still in flight
        } else {
            wait_async_le0();
        }
        __syncthreads();

        // ---- fragments: two contiguous b128 LDS loads each ----
        v16bf afr[2], bfr[4];
#pragma unroll
        for (int mi = 0; mi < 2; ++mi) {
            int row = waveM * 32 + mi * 16 + lrow;
            v8bf lo = *reinterpret_cast<const v8bf*>(&Asm[cur][row][khalf]);
            v8bf hi = *reinterpret_cast<const v8bf*>(&Asm[cur][row][16 + khalf]);
            afr[mi] = __builtin_shufflevector(lo, hi, 0, 1, 2, 3, 4, 5, 6, 7,
                                              8, 9, 10, 11, 12, 13, 14, 15);
        }
#pragma unroll
        for (int ni = 0; ni < 4; ++ni) {
            int col = waveN * 64 + ni * 16 + lrow;
            v8bf lo = *reinterpret_cast<const v8bf*>(&Bsm[cur][col][kb]);
            v8bf hi = *reinterpret_cast<const v8bf*>(&Bsm[cur][col][16 + kb]);
            bfr[ni] = __builtin_shufflevector(lo, hi, 0, 1, 2, 3, 4, 5, 6, 7,
                                              8, 9, 10, 11, 12, 13, 14, 15);
        }

#pragma unroll
        for (int mi = 0; mi < 2; ++mi)
#pragma unroll
            for (int ni = 0; ni < 4; ++ni)
                acc[mi][ni] = __builtin_amdgcn_wmma_f32_16x16x32_bf16(
                    false, afr[mi], false, bfr[ni],
                    (short)0, acc[mi][ni], false, false);
        __syncthreads();   // protect buffer cur before stage s+2 overwrites it
    }

    // ---- epilogue: C/D layout — VGPR e -> M = (lane>=16 ? 8:0)+e, N = lane&15 ----
    const int mhi = (lane >> 4) * 8;
#pragma unroll
    for (int mi = 0; mi < 2; ++mi) {
#pragma unroll
        for (int ni = 0; ni < 4; ++ni) {
            int colg = n0 + waveN * 64 + ni * 16 + lrow;
            float bv = bias[colg];
#pragma unroll
            for (int e = 0; e < 8; ++e) {
                int rowl = waveM * 32 + mi * 16 + mhi + e;
                long gi = (long)(m0 + rowl) * CH + colg;
                float rr = acc[mi][ni][e] + bv;
                if (resid) rr += resid[gi];
                out[gi] = rr;
            }
        }
    }
}

// ---------------------------------------------------------------------------
// Orchestration
// ---------------------------------------------------------------------------
extern "C" void kernel_launch(void* const* d_in, const int* in_sizes, int n_in,
                              void* d_out, int out_size, void* d_ws, size_t ws_size,
                              hipStream_t stream) {
    const float* x      = (const float*)d_in[0];
    const float* v1     = (const float*)d_in[1];
    const float* g1     = (const float*)d_in[2];
    const float* b1     = (const float*)d_in[3];
    const float* v2     = (const float*)d_in[4];
    const float* g2     = (const float*)d_in[5];
    const float* b2     = (const float*)d_in[6];
    const float* alphas = (const float*)d_in[7];
    float* out = (float*)d_out;

    const long KCC = (long)KTAP * CH * CH;            // 786432 per layer
    const long BTC = (long)BATCH * TLEN * CH;         // 16777216

    char* ws = (char*)d_ws;
    __bf16* wbf   = (__bf16*)ws;                                    // 6*KCC bf16
    __bf16* actb  = (__bf16*)(ws + 6 * KCC * 2);                    // BTC bf16
    float*  conv1 = (float*)(ws + 6 * KCC * 2 + BTC * 2);           // BTC f32
    float*  xcur  = (float*)(ws + 6 * KCC * 2 + BTC * 2 + BTC * 4); // BTC f32

    for (int i = 0; i < 3; ++i) {
        wnorm_kernel<<<CH, 256, 0, stream>>>(v1 + i * KCC, g1 + (long)i * CH,
                                             wbf + i * KCC);
        wnorm_kernel<<<CH, 256, 0, stream>>>(v2 + i * KCC, g2 + (long)i * CH,
                                             wbf + (3 + i) * KCC);
    }

    int ablocks = (int)(BTC / 256);
    dim3 cgrid((BATCH * TLEN) / TM, CH / TN);
    const int DIL[3] = {1, 3, 5};

    const float* xsrc = x;
    for (int i = 0; i < 3; ++i) {
        act1d_kernel<<<ablocks, 256, 0, stream>>>(xsrc, alphas + (long)(2 * i) * CH,
                                                  actb);
        conv_wmma_kernel<<<cgrid, 256, 0, stream>>>(actb, wbf + i * KCC,
                                                    b1 + (long)i * CH, nullptr,
                                                    conv1, DIL[i]);
        act1d_kernel<<<ablocks, 256, 0, stream>>>(conv1,
                                                  alphas + (long)(2 * i + 1) * CH,
                                                  actb);
        float* dst = (i == 2) ? out : xcur;
        conv_wmma_kernel<<<cgrid, 256, 0, stream>>>(actb, wbf + (3 + i) * KCC,
                                                    b2 + (long)i * CH, xsrc,
                                                    dst, 1);
        xsrc = dst;
    }
}
